// FlowLSTM_78872779424187
// MI455X (gfx1250) — matrix operand order repaired and checked by reading, not compiled, not run on hardware
//
#include <hip/hip_runtime.h>

// ---------------------------------------------------------------------------
// 2-layer LSTM (B=128, T=512, I=128, H=512) + output projection, for gfx1250.
// bf16 WMMA (v_wmma_f32_16x16x32_bf16), f32 accumulate, f32 cell state.
// - Weights AND the whole x tensor converted to bf16 once in a prep pass
//   (weights ~6.6MB L2-resident; x_bf16 16.7MB), so the sequential inner
//   loops are pure {global_load_b128 -> v_wmma} with no in-loop converts.
// - Host loop over time, 2 kernels/step; all T output projections batched
//   into one final parallel GEMM via a 64MB h2 history buffer.
// - Each wave owns TWO 16x16 output tiles (two accumulators): every B
//   fragment is reused twice and independent WMMAs interleave in the XDL pipe.
// - Deterministic fallbacks if ws_size is too small for the big buffers.
// ---------------------------------------------------------------------------

typedef __attribute__((ext_vector_type(16))) __bf16 v16bf;
typedef __attribute__((ext_vector_type(8)))  __bf16 v8bf;
typedef __attribute__((ext_vector_type(8)))  float  v8f;
typedef __attribute__((ext_vector_type(4)))  float  v4f;

#define B_ 128
#define T_ 512
#define I_ 128
#define H_ 512
#define G_ 2048   // 4*H

__device__ __forceinline__ float sig_f(float x) { return 1.0f / (1.0f + __expf(-x)); }
__device__ __forceinline__ float tanh_f(float x) { return 2.0f * sig_f(2.0f * x) - 1.0f; }

// A-fragment (16x32 bf16, M x K). Lane L: m = L%16, half = L/16.
// elems 0..7  <- A[m][k0 + 8*half + 0..7]
// elems 8..15 <- A[m][k0 + 16 + 8*half + 0..7]
__device__ __forceinline__ v16bf frag_a_bf16(const __bf16* row, int k0, int half) {
  union { v16bf v; v8bf h[2]; } u;
  u.h[0] = *(const v8bf*)(row + k0 + 8 * half);
  u.h[1] = *(const v8bf*)(row + k0 + 16 + 8 * half);
  return u.v;
}

// A-fragment sourced from f32, converted on the fly (fallback path only).
__device__ __forceinline__ v16bf frag_a_f32(const float* row, int k0, int half) {
  const v4f f0 = *(const v4f*)(row + k0 + 8 * half);
  const v4f f1 = *(const v4f*)(row + k0 + 8 * half + 4);
  const v4f f2 = *(const v4f*)(row + k0 + 16 + 8 * half);
  const v4f f3 = *(const v4f*)(row + k0 + 16 + 8 * half + 4);
  v16bf r;
#pragma unroll
  for (int i = 0; i < 4; ++i) {
    r[i]      = (__bf16)f0[i];
    r[i + 4]  = (__bf16)f1[i];
    r[i + 8]  = (__bf16)f2[i];
    r[i + 12] = (__bf16)f3[i];
  }
  return r;
}

// B-fragment (32x16 bf16, K x N) from row-major [N,K] weight:
// lane L holds W[ntile + L%16][k0 + 8*(L/16) + 0..7] and [k0 + 16 + 8*(L/16) + 0..7].
__device__ __forceinline__ v16bf frag_b_bf16(const __bf16* wrow, int k0, int half) {
  union { v16bf v; v8bf h[2]; } u;
  u.h[0] = *(const v8bf*)(wrow + k0 + 8 * half);
  u.h[1] = *(const v8bf*)(wrow + k0 + 16 + 8 * half);
  return u.v;
}

__device__ __forceinline__ v8f wmma_bf16(v16bf a, v16bf b, v8f c) {
  return __builtin_amdgcn_wmma_f32_16x16x32_bf16(false, a, false, b, (short)0, c,
                                                 false, false);
}

// ---------------------------------------------------------------------------
// Prep: weights + x f32->bf16 (layouts unchanged), bias sums, zero states.
// ---------------------------------------------------------------------------
__global__ void lstm_prep(
    const float* __restrict__ x,
    const float* __restrict__ Wih1, const float* __restrict__ Whh1,
    const float* __restrict__ bih1, const float* __restrict__ bhh1,
    const float* __restrict__ Wih2, const float* __restrict__ Whh2,
    const float* __restrict__ bih2, const float* __restrict__ bhh2,
    const float* __restrict__ Wout,
    __bf16* __restrict__ xbf,  // may be null (fallback)
    __bf16* __restrict__ wih1b, __bf16* __restrict__ whh1b,
    __bf16* __restrict__ wih2b, __bf16* __restrict__ whh2b,
    __bf16* __restrict__ woutb,
    float* __restrict__ b1, float* __restrict__ b2,
    float* __restrict__ c1, float* __restrict__ c2,
    __bf16* __restrict__ h1a, __bf16* __restrict__ h1b,
    __bf16* __restrict__ h2a, __bf16* __restrict__ h2b,
    __bf16* __restrict__ zrow)
{
  const size_t tid = (size_t)blockIdx.x * blockDim.x + threadIdx.x;
  const size_t stride = (size_t)gridDim.x * blockDim.x;

  for (size_t k = tid; k < (size_t)G_ * H_; k += stride) {
    whh1b[k] = (__bf16)Whh1[k];
    wih2b[k] = (__bf16)Wih2[k];
    whh2b[k] = (__bf16)Whh2[k];
    if (k < (size_t)G_ * I_) wih1b[k] = (__bf16)Wih1[k];
    if (k < (size_t)I_ * H_) {
      woutb[k] = (__bf16)Wout[k];
      c1[k] = 0.0f; c2[k] = 0.0f;
      h1a[k] = (__bf16)0.0f; h1b[k] = (__bf16)0.0f;
      h2a[k] = (__bf16)0.0f; h2b[k] = (__bf16)0.0f;
    }
    if (k < (size_t)G_) { b1[k] = bih1[k] + bhh1[k]; b2[k] = bih2[k] + bhh2[k]; }
    if (k < (size_t)H_) zrow[k] = (__bf16)0.0f;
  }

  if (xbf) {
    for (size_t k = tid; k < (size_t)B_ * T_ * I_; k += stride)
      xbf[k] = (__bf16)x[k];
  }
}

// ---------------------------------------------------------------------------
// One LSTM layer, one time step.
//   gates = A1 @ Wih^T + A2 @ Whh^T + bias  (A1 = x_t or h1; K1 = 128 or 512)
// Block = 32(batch) x 16(hidden) patch; wave g computes gate g's two 16x16
// tiles (two accumulators, shared B fragment). grid (4, 32), 128 threads.
// xf32 is non-null only in the no-workspace fallback.
// ---------------------------------------------------------------------------
__global__ __launch_bounds__(128) void lstm_layer_step(
    const float* __restrict__ xf32, int t,
    const __bf16* __restrict__ A1, int ldA1, int K1,
    const __bf16* __restrict__ Wih,
    const __bf16* __restrict__ A2, int ldA2,
    const __bf16* __restrict__ Whh,
    const float* __restrict__ bias,
    float* __restrict__ c_state,
    __bf16* __restrict__ h_out, int ld_out)
{
  const int mt   = blockIdx.x;          // batch super-tile 0..3 (32 rows)
  const int ct   = blockIdx.y;          // hidden tile 0..31
  const int gate = threadIdx.x >> 5;    // 0:i 1:f 2:g 3:o
  const int lane = threadIdx.x & 31;
  const int half = lane >> 4;
  const int nl   = lane & 15;
  const int m0   = mt * 32 + nl;        // A row, first M tile
  const int m1   = m0 + 16;             // A row, second M tile
  const int ncol = gate * H_ + ct * 16 + nl;   // gate column == weight row

  v8f acc0 = {}, acc1 = {};

  const __bf16* wih_row = Wih + (size_t)ncol * K1;
  if (xf32) {  // fallback: layer-1 input from f32 x, K = 128
    const float* xr0 = xf32 + ((size_t)m0 * T_ + t) * I_;
    const float* xr1 = xf32 + ((size_t)m1 * T_ + t) * I_;
#pragma unroll
    for (int k0 = 0; k0 < I_; k0 += 32) {
      const v16bf b = frag_b_bf16(wih_row, k0, half);
      acc0 = wmma_bf16(frag_a_f32(xr0, k0, half), b, acc0);
      acc1 = wmma_bf16(frag_a_f32(xr1, k0, half), b, acc1);
    }
  } else {  // input contribution from bf16 A1 (x_bf16 or h1), K = K1
    const __bf16* a0 = A1 + (size_t)m0 * ldA1;
    const __bf16* a1 = A1 + (size_t)m1 * ldA1;
#pragma unroll 2
    for (int k0 = 0; k0 < K1; k0 += 32) {
      const v16bf b = frag_b_bf16(wih_row, k0, half);
      acc0 = wmma_bf16(frag_a_bf16(a0, k0, half), b, acc0);
      acc1 = wmma_bf16(frag_a_bf16(a1, k0, half), b, acc1);
    }
  }
  {  // recurrent contribution, K = 512
    const __bf16* h0 = A2 + (size_t)m0 * ldA2;
    const __bf16* h1 = A2 + (size_t)m1 * ldA2;
    const __bf16* whh_row = Whh + (size_t)ncol * H_;
#pragma unroll 2
    for (int k0 = 0; k0 < H_; k0 += 32) {
      const v16bf b = frag_b_bf16(whh_row, k0, half);
      acc0 = wmma_bf16(frag_a_bf16(h0, k0, half), b, acc0);
      acc1 = wmma_bf16(frag_a_bf16(h1, k0, half), b, acc1);
    }
  }

  // Activated gates -> LDS. C/D layout: VGPR r holds (M = r + 8*half, N = nl).
  __shared__ float act[4][32][16];
  const float bsum = bias[ncol];
#pragma unroll
  for (int r = 0; r < 8; ++r) {
    float v0 = acc0[r] + bsum;
    float v1 = acc1[r] + bsum;
    if (gate == 2) { v0 = tanh_f(v0); v1 = tanh_f(v1); }
    else           { v0 = sig_f(v0);  v1 = sig_f(v1);  }
    act[gate][r + 8 * half][nl]      = v0;
    act[gate][16 + r + 8 * half][nl] = v1;
  }
  __syncthreads();

  // Fused cell update: 128 threads cover the 512-element patch.
  for (int idx = threadIdx.x; idx < 512; idx += 128) {
    const int ml = idx >> 4, nn = idx & 15;
    const int gm = mt * 32 + ml, gn = ct * 16 + nn;
    float* cp = c_state + (size_t)gm * H_ + gn;
    const float cold = *cp;
    const float cnew = act[1][ml][nn] * cold + act[0][ml][nn] * act[2][ml][nn];
    const float hnew = act[3][ml][nn] * tanh_f(cnew);
    *cp = cnew;
    h_out[(size_t)gm * ld_out + gn] = (__bf16)hnew;
  }
}

// ---------------------------------------------------------------------------
// Batched projection over the whole sequence:
//   out[r, :] = H2all[r, :] @ Wout^T + b_out,  r = b*T + t  (== [B,T,I] order)
// Block covers 128 rows x 16 cols; wave w owns rows w*32..w*32+31 (2 tiles).
// grid (512, 8), 128 threads.
// ---------------------------------------------------------------------------
__global__ __launch_bounds__(128) void out_proj_all(
    const __bf16* __restrict__ H2all, const __bf16* __restrict__ Wout,
    const float* __restrict__ b_out, float* __restrict__ out)
{
  const int rt = blockIdx.x;            // row group of 128
  const int nt = blockIdx.y;            // 0..7
  const int wv = threadIdx.x >> 5;
  const int lane = threadIdx.x & 31;
  const int half = lane >> 4, nl = lane & 15;
  const int r0 = rt * 128 + wv * 32 + nl;
  const int r1 = r0 + 16;
  const int n  = nt * 16 + nl;

  v8f acc0 = {}, acc1 = {};
  const __bf16* a0 = H2all + (size_t)r0 * H_;
  const __bf16* a1 = H2all + (size_t)r1 * H_;
  const __bf16* wrow = Wout + (size_t)n * H_;
#pragma unroll 2
  for (int k0 = 0; k0 < H_; k0 += 32) {
    const v16bf b = frag_b_bf16(wrow, k0, half);
    acc0 = wmma_bf16(frag_a_bf16(a0, k0, half), b, acc0);
    acc1 = wmma_bf16(frag_a_bf16(a1, k0, half), b, acc1);
  }
  const float bn = b_out[n];
#pragma unroll
  for (int r = 0; r < 8; ++r) {
    const int row0 = rt * 128 + wv * 32 + r + 8 * half;
    out[(size_t)row0 * I_ + n]        = acc0[r] + bn;
    out[(size_t)(row0 + 16) * I_ + n] = acc1[r] + bn;
  }
}

// ---------------------------------------------------------------------------
// Fallback per-step projection (used only when ws is too small for H2all).
// ---------------------------------------------------------------------------
__global__ __launch_bounds__(32) void out_proj_step(
    const __bf16* __restrict__ h2, const __bf16* __restrict__ Wout,
    const float* __restrict__ b_out, float* __restrict__ out, int t)
{
  const int mt = blockIdx.x, nt = blockIdx.y;
  const int lane = threadIdx.x & 31;
  const int half = lane >> 4, nl = lane & 15;
  const int m = mt * 16 + nl;
  const int n = nt * 16 + nl;

  v8f acc = {};
  const __bf16* hrow = h2 + (size_t)m * H_;
  const __bf16* wrow = Wout + (size_t)n * H_;
#pragma unroll 2
  for (int k0 = 0; k0 < H_; k0 += 32)
    acc = wmma_bf16(frag_a_bf16(hrow, k0, half), frag_b_bf16(wrow, k0, half), acc);

  const float bn = b_out[n];
#pragma unroll
  for (int r = 0; r < 8; ++r) {
    const int gm = mt * 16 + r + 8 * half;
    out[((size_t)gm * T_ + t) * I_ + n] = acc[r] + bn;
  }
}

// ---------------------------------------------------------------------------
extern "C" void kernel_launch(void* const* d_in, const int* in_sizes, int n_in,
                              void* d_out, int out_size, void* d_ws, size_t ws_size,
                              hipStream_t stream) {
  (void)in_sizes; (void)n_in; (void)out_size;

  const float* x    = (const float*)d_in[0];
  const float* Wih1 = (const float*)d_in[1];
  const float* Whh1 = (const float*)d_in[2];
  const float* bih1 = (const float*)d_in[3];
  const float* bhh1 = (const float*)d_in[4];
  const float* Wih2 = (const float*)d_in[5];
  const float* Whh2 = (const float*)d_in[6];
  const float* bih2 = (const float*)d_in[7];
  const float* bhh2 = (const float*)d_in[8];
  const float* Wout = (const float*)d_in[9];
  const float* bout = (const float*)d_in[10];
  float* out = (float*)d_out;

  // Carve workspace, 256B-aligned slices. Small region ~8MB.
  char* p = (char*)d_ws;
  auto take = [&](size_t bytes) {
    char* r = p;
    p += (bytes + 255) & ~(size_t)255;
    return r;
  };
  __bf16* wih1b = (__bf16*)take((size_t)G_ * I_ * 2);
  __bf16* whh1b = (__bf16*)take((size_t)G_ * H_ * 2);
  __bf16* wih2b = (__bf16*)take((size_t)G_ * H_ * 2);
  __bf16* whh2b = (__bf16*)take((size_t)G_ * H_ * 2);
  __bf16* woutb = (__bf16*)take((size_t)I_ * H_ * 2);
  float*  b1    = (float*)take((size_t)G_ * 4);
  float*  b2    = (float*)take((size_t)G_ * 4);
  float*  c1    = (float*)take((size_t)B_ * H_ * 4);
  float*  c2    = (float*)take((size_t)B_ * H_ * 4);
  __bf16* h1a   = (__bf16*)take((size_t)B_ * H_ * 2);
  __bf16* h1b   = (__bf16*)take((size_t)B_ * H_ * 2);
  __bf16* h2a   = (__bf16*)take((size_t)B_ * H_ * 2);
  __bf16* h2b   = (__bf16*)take((size_t)B_ * H_ * 2);
  __bf16* zrow  = (__bf16*)take((size_t)H_ * 2);

  // Optional big buffers (priority: x_bf16 16.7MB, then h2 history 64MB).
  const size_t xbf_bytes = (size_t)B_ * T_ * I_ * 2;
  const size_t h2all_bytes = (size_t)B_ * T_ * H_ * 2;
  __bf16* xbf = nullptr;
  __bf16* H2all = nullptr;
  if ((size_t)(p - (char*)d_ws) + xbf_bytes <= ws_size)
    xbf = (__bf16*)take(xbf_bytes);
  if ((size_t)(p - (char*)d_ws) + h2all_bytes <= ws_size)
    H2all = (__bf16*)take(h2all_bytes);

  lstm_prep<<<2048, 256, 0, stream>>>(x, Wih1, Whh1, bih1, bhh1, Wih2, Whh2,
                                      bih2, bhh2, Wout, xbf,
                                      wih1b, whh1b, wih2b, whh2b, woutb,
                                      b1, b2, c1, c2, h1a, h1b, h2a, h2b, zrow);

  for (int t = 0; t < T_; ++t) {
    __bf16* h1_rd = (t & 1) ? h1b : h1a;
    __bf16* h1_wr = (t & 1) ? h1a : h1b;

    // layer 1: gates1 = x_t@Wih1^T + h1_prev@Whh1^T + b1
    if (xbf) {
      lstm_layer_step<<<dim3(4, 32), 128, 0, stream>>>(
          (const float*)nullptr, t, xbf + (size_t)t * I_, T_ * I_, I_, wih1b,
          h1_rd, H_, whh1b, b1, c1, h1_wr, H_);
    } else {
      lstm_layer_step<<<dim3(4, 32), 128, 0, stream>>>(
          x, t, (const __bf16*)nullptr, 0, I_, wih1b, h1_rd, H_, whh1b, b1, c1,
          h1_wr, H_);
    }

    // layer 2: gates2 = h1@Wih2^T + h2_prev@Whh2^T + b2
    if (H2all) {
      // h2 history: step t writes slot t, reads slot t-1 (t=0 reads zero row).
      const __bf16* a2 = t ? (H2all + (size_t)(t - 1) * H_) : zrow;
      const int lda2 = t ? (T_ * H_) : 0;
      lstm_layer_step<<<dim3(4, 32), 128, 0, stream>>>(
          (const float*)nullptr, t, h1_wr, H_, H_, wih2b, a2, lda2, whh2b, b2, c2,
          H2all + (size_t)t * H_, T_ * H_);
    } else {
      __bf16* h2_rd = (t & 1) ? h2b : h2a;
      __bf16* h2_wr = (t & 1) ? h2a : h2b;
      lstm_layer_step<<<dim3(4, 32), 128, 0, stream>>>(
          (const float*)nullptr, t, h1_wr, H_, H_, wih2b, h2_rd, H_, whh2b, b2, c2,
          h2_wr, H_);
      out_proj_step<<<dim3(8, 8), 32, 0, stream>>>(h2_wr, woutb, bout, out, t);
    }
  }

  if (H2all)  // one big parallel GEMM: [B*T, H] x [H, I]
    out_proj_all<<<dim3((B_ * T_) / 128, I_ / 16), 128, 0, stream>>>(H2all, woutb,
                                                                     bout, out);
}
